// GATEncoder_15556371546816
// MI455X (gfx1250) — matrix operand order repaired and checked by reading, not compiled, not run on hardware
//
#include <hip/hip_runtime.h>
#include <hip/hip_bf16.h>

typedef __attribute__((ext_vector_type(16))) __bf16 v16bf;
typedef __attribute__((ext_vector_type(8)))  float  v8f;

#define KK 512
#define GAT_ALPHA 0.2f
#define GAT_NEG_MASK -9.0e15f

__device__ __forceinline__ __bf16 f2bf(float f) { return (__bf16)f; }   // fptrunc -> native cvt
__device__ __forceinline__ float  bf2f(__bf16 b) { return (float)b; }

// -----------------------------------------------------------------------------
// Wh^T[b][col][k] (bf16) = (A[b*KK+k][:] @ W[:][col]) ; one wave per 16x16 tile.
// A is f32 row-major [B*KK x Kd]; Kd multiple of 32 (224, 64). N = #cols (64/32).
// Output stored K-major so the attention B-fragment is a contiguous 32B load.
// -----------------------------------------------------------------------------
__global__ void __launch_bounds__(32)
gat_gemm_to_bf16T(const float* __restrict__ A, const float* __restrict__ W,
                  __bf16* __restrict__ WhT, int Kd, int N) {
  const int lane   = threadIdx.x;
  const int m      = lane & 15;      // A row / B col within tile
  const int half   = lane >> 4;      // 0|1
  const int khalfA = half * 16;      // A k order used here: K = khalf+0..15
  const int koffB  = half * 16;      // B k order used here (same as A): K = koff+0..15
  const int rowBase = blockIdx.x * 16;
  const int colBase = blockIdx.y * 16;
  const float* Arow = A + (size_t)(rowBase + m) * Kd + khalfA;

  v8f acc = {};
  for (int k0 = 0; k0 < Kd; k0 += 32) {
    float4 a0 = *(const float4*)(Arow + k0);
    float4 a1 = *(const float4*)(Arow + k0 + 4);
    float4 a2 = *(const float4*)(Arow + k0 + 8);
    float4 a3 = *(const float4*)(Arow + k0 + 12);
    v16bf a, b;
    a[0] = f2bf(a0.x); a[1] = f2bf(a0.y); a[2]  = f2bf(a0.z); a[3]  = f2bf(a0.w);
    a[4] = f2bf(a1.x); a[5] = f2bf(a1.y); a[6]  = f2bf(a1.z); a[7]  = f2bf(a1.w);
    a[8] = f2bf(a2.x); a[9] = f2bf(a2.y); a[10] = f2bf(a2.z); a[11] = f2bf(a2.w);
    a[12]= f2bf(a3.x); a[13]= f2bf(a3.y); a[14] = f2bf(a3.z); a[15] = f2bf(a3.w);
#pragma unroll
    for (int j = 0; j < 16; ++j)
      b[j] = f2bf(W[(size_t)(k0 + koffB + j) * N + colBase + m]);   // tiny, L1-resident
    acc = __builtin_amdgcn_wmma_f32_16x16x32_bf16(false, a, false, b,
                                                  (short)0, acc, false, false);
  }
#pragma unroll
  for (int r = 0; r < 8; ++r) {                  // C/D layout: M = half*8 + r
    int row = rowBase + half * 8 + r;            // global row in [0, B*KK)
    int bq = row >> 9, rk = row & (KK - 1);
    WhT[(size_t)bq * N * KK + (size_t)(colBase + m) * KK + rk] = f2bf(acc[r]);
  }
}

// f_src[r] = Wh[r,:] . a[0:F] ; f_dst[r] = Wh[r,:] . a[F:2F]  (Wh read from bf16 Wh^T)
__global__ void gat_fvec(const __bf16* __restrict__ WhT, const float* __restrict__ avec,
                         float* __restrict__ fsrc, float* __restrict__ fdst,
                         int F, int nrows) {
  int r = blockIdx.x * blockDim.x + threadIdx.x;
  if (r >= nrows) return;
  int bq = r >> 9, rk = r & (KK - 1);
  const __bf16* base = WhT + (size_t)bq * F * KK + rk;
  float s = 0.f, d = 0.f;
  for (int f = 0; f < F; ++f) {
    float w = bf2f(base[(size_t)f * KK]);        // coalesced across threads (rk)
    s += w * avec[f]; d += w * avec[F + f];
  }
  fsrc[r] = s; fdst[r] = d;
}

// -----------------------------------------------------------------------------
// One block = (batch bb, 16-row tile).
// Phase 1: e = leakyrelu(f_src[i]+f_dst[j]) masked by adj (the single coalesced
//          pass over adj = the HBM-roofline term), row softmax -> bf16 LDS.
// Phase 2: out[16 x F] = att @ Wh via WMMA (A: 2x16B LDS chunks, B: 32B global
//          vector load from bf16 Wh^T), then ELU.
// -----------------------------------------------------------------------------
__global__ void __launch_bounds__(256)
gat_attn(const __bf16* __restrict__ WhT, const float* __restrict__ fsrc,
         const float* __restrict__ fdst, const int* __restrict__ adj,
         float* __restrict__ out, int F) {
  __shared__ __bf16 sAtt[16 * KK];   // 16 KB: softmax probabilities (WMMA A)
  __shared__ float  sRed[16 * 16];

  const int tid  = threadIdx.x;
  const int bb   = blockIdx.y;
  const int tile = blockIdx.x;
  const int i    = tid >> 4;         // tile row 0..15
  const int t16  = tid & 15;         // 32-column chunk id
  const int gi   = tile * 16 + i;
  const int jbase = t16 * 32;

  const float fs = fsrc[bb * KK + gi];
  const float* fdRow  = fdst + (size_t)bb * KK;
  const int*   adjRow = adj + (size_t)bb * KK * KK + (size_t)gi * KK;

  float sc[32];
  float mx = GAT_NEG_MASK;
#pragma unroll
  for (int c = 0; c < 8; ++c) {
    int j = jbase + c * 4;
    int4   ad  = *(const int4*)(adjRow + j);     // b128 adj loads
    float4 fd4 = *(const float4*)(fdRow + j);
    float e;
    e = fs + fd4.x; e = e > 0.f ? e : GAT_ALPHA * e; e = ad.x > 0 ? e : GAT_NEG_MASK;
    sc[c*4+0] = e; mx = fmaxf(mx, e);
    e = fs + fd4.y; e = e > 0.f ? e : GAT_ALPHA * e; e = ad.y > 0 ? e : GAT_NEG_MASK;
    sc[c*4+1] = e; mx = fmaxf(mx, e);
    e = fs + fd4.z; e = e > 0.f ? e : GAT_ALPHA * e; e = ad.z > 0 ? e : GAT_NEG_MASK;
    sc[c*4+2] = e; mx = fmaxf(mx, e);
    e = fs + fd4.w; e = e > 0.f ? e : GAT_ALPHA * e; e = ad.w > 0 ? e : GAT_NEG_MASK;
    sc[c*4+3] = e; mx = fmaxf(mx, e);
  }
  sRed[i * 16 + t16] = mx;
  __syncthreads();
  float rmx = sRed[i * 16];
  for (int q = 1; q < 16; ++q) rmx = fmaxf(rmx, sRed[i * 16 + q]);
  __syncthreads();
  float sum = 0.f;
#pragma unroll 4
  for (int c = 0; c < 32; ++c) { sc[c] = expf(sc[c] - rmx); sum += sc[c]; }
  sRed[i * 16 + t16] = sum;
  __syncthreads();
  float rsum = 0.f;
  for (int q = 0; q < 16; ++q) rsum += sRed[i * 16 + q];
  const float inv = 1.f / rsum;
#pragma unroll
  for (int c = 0; c < 32; ++c)                    // contiguous -> merged ds stores
    sAtt[i * KK + jbase + c] = f2bf(sc[c] * inv);
  __syncthreads();

  // WMMA phase: wave w -> output cols [w*16, (w+1)*16). Uniform branch (EXEC all-1s).
  const int wave = tid >> 5;
  const int lane = tid & 31;
  const int ntiles = F >> 4;
  if (wave < ntiles) {
    const int m      = lane & 15;
    const int half   = lane >> 4;
    const int khalfA = half * 16;
    const int koffB  = half * 16;
    const int col0   = wave * 16;
    const __bf16* aRow = sAtt + m * KK + khalfA;
    const __bf16* bCol = WhT + (size_t)bb * F * KK + (size_t)(col0 + m) * KK + koffB;

    v8f acc = {};
#pragma unroll 4
    for (int k0 = 0; k0 < KK; k0 += 32) {
      v16bf a, b;
      __builtin_memcpy(&a, aRow + k0, 16);                 // K khalf+0..7
      __builtin_memcpy((char*)&a + 16, aRow + k0 + 8, 16); // K khalf+8..15
      __builtin_memcpy(&b, bCol + k0, 32);                 // K koff+0..15 contiguous
      acc = __builtin_amdgcn_wmma_f32_16x16x32_bf16(false, a, false, b,
                                                    (short)0, acc, false, false);
    }
#pragma unroll
    for (int r = 0; r < 8; ++r) {
      int row = tile * 16 + half * 8 + r;
      float v = acc[r];
      v = v > 0.f ? v : expm1f(v);                          // elu(alpha=1)
      out[((size_t)bb * KK + row) * F + col0 + m] = v;
    }
  }
}

extern "C" void kernel_launch(void* const* d_in, const int* in_sizes, int n_in,
                              void* d_out, int out_size, void* d_ws, size_t ws_size,
                              hipStream_t stream) {
  const float* x  = (const float*)d_in[0];   // (128,512,224)
  const int*   adj= (const int*)  d_in[1];   // (128,512,512)
  const float* W1 = (const float*)d_in[2];   // (224,64)
  const float* a1 = (const float*)d_in[3];   // (128,)
  const float* W2 = (const float*)d_in[4];   // (64,32)
  const float* a2 = (const float*)d_in[5];   // (64,)
  float* outp = (float*)d_out;               // (128,512,32)

  const int B = 128, K = 512, Fin = 224, H = 64, Fout = 32;
  const size_t BK = (size_t)B * K;

  char* ws = (char*)d_ws;
  __bf16* WhT1 = (__bf16*)(ws);                                  // B*H*K bf16 (8 MB)
  float*  H1   = (float*)(ws + (size_t)B * H * K * sizeof(__bf16)); // BK*H f32 (16 MB)
  float*  fs1  = (float*)(ws + (size_t)B * H * K * sizeof(__bf16)
                             + BK * H * sizeof(float));
  float*  fd1  = fs1 + BK;
  __bf16* WhT2 = WhT1;     // layer-2 scratch aliases (WhT1 dead after first attn)
  float*  fs2  = fs1;
  float*  fd2  = fd1;

  // ---- Layer 1 ----
  gat_gemm_to_bf16T<<<dim3(B * K / 16, H / 16), 32, 0, stream>>>(x, W1, WhT1, Fin, H);
  gat_fvec<<<(int)((BK + 255) / 256), 256, 0, stream>>>(WhT1, a1, fs1, fd1, H, (int)BK);
  gat_attn<<<dim3(K / 16, B), 256, 0, stream>>>(WhT1, fs1, fd1, adj, H1, H);

  // ---- Layer 2 ----
  gat_gemm_to_bf16T<<<dim3(B * K / 16, Fout / 16), 32, 0, stream>>>(H1, W2, WhT2, H, Fout);
  gat_fvec<<<(int)((BK + 255) / 256), 256, 0, stream>>>(WhT2, a2, fs2, fd2, Fout, (int)BK);
  gat_attn<<<dim3(K / 16, B), 256, 0, stream>>>(WhT2, fs2, fd2, adj, outp, Fout);
}
